// Encoder_Postnet_31095563223393
// MI455X (gfx1250) — matrix operand order repaired and checked by reading, not compiled, not run on hardware
//
#include <hip/hip_runtime.h>
#include <hip/hip_bf16.h>
#include <cmath>

// ---------------------------------------------------------------------------
// Shapes (fixed by the reference)
// ---------------------------------------------------------------------------
#define BATCH   16
#define T_LEN   8192
#define TP_LEN  512
#define E_DIM   256

// GEMM tiling
#define MB      64          // rows of A per block
#define LDA     264         // padded LDS row stride in bf16 elems (132 dwords == 4 mod 64 -> conflict free)

typedef __attribute__((ext_vector_type(16))) __bf16 v16bf;
typedef __attribute__((ext_vector_type(8)))  __bf16 v8bf;
typedef __attribute__((ext_vector_type(4)))  __bf16 v4bf;
typedef __attribute__((ext_vector_type(8)))  float  v8f;

// ---------------------------------------------------------------------------
// Kernel 1: sinusoid positional encoding, f32, (T, E).  Lives in L2 (8 MB).
// ---------------------------------------------------------------------------
__global__ __launch_bounds__(256) void pe_kernel(float* __restrict__ pe) {
    const int t = blockIdx.x;          // 0..8191
    const int e = threadIdx.x;         // 0..255
    const int i = e >> 1;              // div index
    // div[i] = exp(-i * ln(10000)/128)
    const float ang = (float)t * __expf((float)i * -0.07195588780e0f);
    pe[(size_t)t * E_DIM + e] = (e & 1) ? cosf(ang) : sinf(ang);
}

// ---------------------------------------------------------------------------
// Kernel 2: pack W_pos (E x E, row-major f32) into bf16 WMMA-B fragments.
// B layout used here (16-bit, 32x16, contiguous K, not ISA table order): lanes 0-15 hold K=0..15 of column
// (lane&15); lanes 16-31 hold K=16..31.  Element j of the 16-elem per-lane
// vector <-> K = 16*(lane>>4) + j.
// Fragment (ntile, kstep) occupies 32 lanes * 16 bf16 contiguous.
// ---------------------------------------------------------------------------
__global__ __launch_bounds__(256) void wpack_kernel(const float* __restrict__ W,
                                                    __bf16* __restrict__ wpack) {
    const int v    = blockIdx.x * 256 + threadIdx.x;   // 0..65535
    const int j    = v & 15;
    const int lane = (v >> 4) & 31;
    const int ks   = (v >> 9) & 7;
    const int nt   = v >> 12;
    const int k    = ks * 32 + ((lane >> 4) * 16 + j);
    const int n    = nt * 16 + (lane & 15);
    wpack[v] = (__bf16)W[k * E_DIM + n];
}

// ---------------------------------------------------------------------------
// Kernel 3: aligner index scan.  Invariant: before == text[ind], so the state
// is a single int.  Sequential along T (true data dependence), parallel over
// batches: one wave per batch; text row cached in LDS so the dependent lookup
// is an LDS hit; align loads are state-independent and pipeline.
// ---------------------------------------------------------------------------
__global__ __launch_bounds__(32) void scan_kernel(const int* __restrict__ align,
                                                  const int* __restrict__ text,
                                                  int* __restrict__ idxout) {
    __shared__ int st[TP_LEN];
    const int b = blockIdx.x;
    for (int i = threadIdx.x; i < TP_LEN; i += 32) st[i] = text[b * TP_LEN + i];
    __syncthreads();
    if (threadIdx.x == 0) {
        const int* ap = align + (size_t)b * T_LEN;
        int*       op = idxout + (size_t)b * T_LEN;
        int ind = 0, before = st[0];
        op[0] = 0;
        #pragma unroll 4
        for (int jj = 1; jj < T_LEN; ++jj) {
            const int a = ap[jj];
            if (a != before) { ind = min(ind + 1, TP_LEN - 1); before = st[ind]; }
            op[jj] = ind;
        }
    }
}

// ---------------------------------------------------------------------------
// Kernel 4: fused  out = gather + pitch*Wp + bp + emb[beats] + b_pos
//                       + (bf16(gather + pe) @ bf16(W_pos))   [f32 accum]
// Block: 64 rows x full N=256.  8 waves; wave w owns N-tiles {2w, 2w+1}.
// ---------------------------------------------------------------------------
__global__ __launch_bounds__(256) void gemm_fused(
    const float* __restrict__ enc,      // (B, TP, E)
    const float* __restrict__ pitch,    // (B*T)
    const int*   __restrict__ beats,    // (B*T)
    const float* __restrict__ Wpitch,   // (E)
    const float* __restrict__ bpitch,   // (E)
    const float* __restrict__ bpos,     // (E)
    const float* __restrict__ embb,     // (2, E)
    const int*   __restrict__ idxb,     // (B*T)
    const __bf16* __restrict__ wpack,   // packed B fragments
    const float* __restrict__ pe,       // (T, E)
    float* __restrict__ out)            // (B*T, E)
{
    __shared__ __bf16 As[MB * LDA];

    const int tid  = threadIdx.x;
    const int row0 = blockIdx.x * MB;          // tiles never straddle batches (8192 % 64 == 0)
    const int bb   = row0 >> 13;               // batch index

    // ---- stage A = bf16(enc[idx] + pe) into LDS (float4-vectorized) ----
    #pragma unroll
    for (int it = 0; it < 16; ++it) {
        const int c  = it * 256 + tid;         // float4 chunk id, 0..4095
        const int r  = c >> 6;                 // local row 0..63
        const int c4 = (c & 63) << 2;          // float column (16B aligned)
        const int gm = row0 + r;
        const int t  = gm & (T_LEN - 1);
        const int id = idxb[gm];
        const float4 ev = *(const float4*)(enc + ((size_t)(bb * TP_LEN + id)) * E_DIM + c4);
        const float4 pv = *(const float4*)(pe + (size_t)t * E_DIM + c4);
        v4bf sv;
        sv[0] = (__bf16)(ev.x + pv.x);
        sv[1] = (__bf16)(ev.y + pv.y);
        sv[2] = (__bf16)(ev.z + pv.z);
        sv[3] = (__bf16)(ev.w + pv.w);
        *(v4bf*)&As[r * LDA + c4] = sv;
    }
    __syncthreads();

    const int lane = tid & 31;
    const int wave = tid >> 5;
    const int nt0  = wave * 2;                 // this wave's first N-tile
    const int half = lane >> 4;                // lane half (K-subset select)
    const int m    = lane & 15;                // row within M-tile / col within N-tile

    v8f acc[4][2];
    #pragma unroll
    for (int mt = 0; mt < 4; ++mt) {
        acc[mt][0] = (v8f){0.f,0.f,0.f,0.f,0.f,0.f,0.f,0.f};
        acc[mt][1] = (v8f){0.f,0.f,0.f,0.f,0.f,0.f,0.f,0.f};
    }

    // ---- K loop: 8 steps of K=32, v_wmma_f32_16x16x32_bf16 ----
    #pragma unroll
    for (int ks = 0; ks < 8; ++ks) {
        const int k0 = ks * 32;
        // B fragments: 32 contiguous bytes per lane (stays hot in L2)
        const v16bf bf0 = *(const v16bf*)(wpack + ((size_t)(((nt0 + 0) * 8 + ks) * 32 + lane)) * 16);
        const v16bf bf1 = *(const v16bf*)(wpack + ((size_t)(((nt0 + 1) * 8 + ks) * 32 + lane)) * 16);
        #pragma unroll
        for (int mt = 0; mt < 4; ++mt) {
            // A fragment: lanes 0-15 K 0..15, lanes 16-31 K 16..31 (matches B packing)
            const __bf16* ap = &As[(mt * 16 + m) * LDA + k0 + half * 16];
            const v8bf lo = *(const v8bf*)ap;
            const v8bf hi = *(const v8bf*)(ap + 8);
            v16bf af;
            #pragma unroll
            for (int i = 0; i < 8; ++i) { af[i] = lo[i]; af[8 + i] = hi[i]; }
            acc[mt][0] = __builtin_amdgcn_wmma_f32_16x16x32_bf16(
                false, af, false, bf0, (short)0, acc[mt][0], false, false);
            acc[mt][1] = __builtin_amdgcn_wmma_f32_16x16x32_bf16(
                false, af, false, bf1, (short)0, acc[mt][1], false, false);
        }
    }

    // ---- fused epilogue.  C/D layout: VGPR j -> M = j + 8*half, N = lane&15 ----
    #pragma unroll
    for (int nti = 0; nti < 2; ++nti) {
        const int n  = (nt0 + nti) * 16 + m;
        const float wn  = Wpitch[n];
        const float bpn = bpitch[n];
        const float bqn = bpos[n];
        const float e0  = embb[n];
        const float e1  = embb[E_DIM + n];
        #pragma unroll
        for (int mt = 0; mt < 4; ++mt) {
            const v8f a = acc[mt][nti];
            #pragma unroll
            for (int j = 0; j < 8; ++j) {
                const int gm = row0 + mt * 16 + j + half * 8;
                const int id = idxb[gm];
                const float encv = enc[((size_t)(bb * TP_LEN + id)) * E_DIM + n];
                const float pv   = pitch[gm];
                const float emb  = beats[gm] ? e1 : e0;
                out[(size_t)gm * E_DIM + n] = encv + (pv * wn + bpn) + emb + (a[j] + bqn);
            }
        }
    }
}

// ---------------------------------------------------------------------------
// Workspace layout (needs ~8.7 MB):
//   [0, 512K)              idx          int32  B*T
//   [512K, 640K)           wpack        bf16   16*8*32*16
//   [640K, 640K + 8M)      pe           f32    T*E
// ---------------------------------------------------------------------------
extern "C" void kernel_launch(void* const* d_in, const int* in_sizes, int n_in,
                              void* d_out, int out_size, void* d_ws, size_t ws_size,
                              hipStream_t stream) {
    const float* enc    = (const float*)d_in[0];
    const int*   align  = (const int*)  d_in[1];
    const int*   text   = (const int*)  d_in[2];
    const float* pitch  = (const float*)d_in[3];
    const int*   beats  = (const int*)  d_in[4];
    const float* Wpitch = (const float*)d_in[5];
    const float* bpitch = (const float*)d_in[6];
    const float* Wpos   = (const float*)d_in[7];
    const float* bposv  = (const float*)d_in[8];
    const float* embb   = (const float*)d_in[9];
    float*       outp   = (float*)d_out;

    char* ws = (char*)d_ws;
    int*    ws_idx = (int*)ws;                          // 524288 B
    __bf16* ws_wp  = (__bf16*)(ws + 524288);            // 131072 B
    float*  ws_pe  = (float*)(ws + 655360);             // 8 MB

    pe_kernel   <<<T_LEN, 256, 0, stream>>>(ws_pe);
    wpack_kernel<<<256,   256, 0, stream>>>(Wpos, ws_wp);
    scan_kernel <<<BATCH, 32,  0, stream>>>(align, text, ws_idx);
    gemm_fused  <<<(BATCH * T_LEN) / MB, 256, 0, stream>>>(
        enc, pitch, beats, Wpitch, bpitch, bposv, embb,
        ws_idx, ws_wp, ws_pe, outp);
}
